// LaCTSWIGLUSelfAttention_56023553409286
// MI455X (gfx1250) — compile-verified, matrix operand order checked by reading, not run on hardware
//
#include <hip/hip_runtime.h>
#include <hip/hip_bf16.h>
#include <math.h>

// ---------------- problem constants (fixed by the reference) ----------------
#define BH   8
#define TSEQ 4096
#define DD   512          // fw head dim
#define DI   512          // intermediate dim
#define CS   256          // chunk size
#define NCH  15           // chunks with weight updates (last chunk forward-only)
#define LD   512          // the ONLY leading dimension appearing in any GEMM here

static const long MSZ = (long)DI * DD;   // 262144 elems, one 512x512 matrix
static const long CSZ = (long)CS * DI;   // 131072 elems, one 256x512 activation

// Newton-Schulz coefficients (5 steps)
static const float NSC[5][3] = {
    {4.0848f, -6.8946f, 2.9270f},
    {3.9505f, -6.3029f, 2.6377f},
    {3.7418f, -5.5913f, 2.3037f},
    {2.8769f, -3.1427f, 1.2046f},
    {2.8366f, -3.0525f, 1.2012f}};

// ---------------- WMMA types ----------------
typedef __attribute__((ext_vector_type(16))) __bf16 v16bf;
typedef __attribute__((ext_vector_type(8)))  float  v8f;

// round-to-nearest-even bf16 lives in the TOP half of the result dword
__device__ __forceinline__ unsigned int rnd_bf(float f) {
    unsigned int u = __float_as_uint(f);
    return u + 0x7FFFu + ((u >> 16) & 1u);
}

// pack two rounded bf16 (top halves of ulo/uhi) into one dword: 1x v_perm_b32
__device__ __forceinline__ unsigned int pk2bf(unsigned int ulo, unsigned int uhi) {
    return __builtin_amdgcn_perm(uhi, ulo, 0x07060302u);
}

// pack 4 floats -> 4 bf16 in one 64-bit value (no 16-bit register moves)
__device__ __forceinline__ unsigned long long pk4bf(float x, float y, float z, float w) {
    unsigned int lo = pk2bf(rnd_bf(x), rnd_bf(y));
    unsigned int hi = pk2bf(rnd_bf(z), rnd_bf(w));
    return ((unsigned long long)hi << 32) | lo;
}

// single bf16 value in the LOW half (for 16-bit LDS stores; store ignores high bits)
__device__ __forceinline__ unsigned short f2bf(float f) {
    return (unsigned short)(rnd_bf(f) >> 16);
}

// WMMA 16-bit fragment slot for K index kk (0..31):
//   lane group h = (kk>>3)&1 ; element e = (kk&7) + 8*((kk>>4)&1) ; pos = h*16 + e
// Permutes only at 8-granularity => aligned quads of kk stay contiguous.
__device__ __forceinline__ int fragslot(int kk) {
    return (((kk >> 3) & 1) << 4) + (kk & 7) + (((kk >> 4) & 1) << 3);
}

__device__ __forceinline__ float sigm(float x) { return 1.0f / (1.0f + __expf(-x)); }

// ---------------------------------------------------------------------------
// Batched GEMM, layout-specialized:  Cout[b] = alpha*op(A)@op(B) + ebeta*Cin[b]
//   AMODE 0: opA[m,k] = A[m*LD + k]   (k unit-stride)
//   AMODE 1: opA[m,k] = A[k*LD + m]   (m unit-stride, i.e. transposed grads)
//   BMODE 0: opB[k,n] = B[n*LD + k]   (k unit-stride, i.e. B^T / weight rows)
//   BMODE 1: opB[k,n] = B[k*LD + n]   (n unit-stride, row-major B)
// C row-major, ld N.  ebeta = beta * (*beta_dev if given).
// bf16 staged in LDS in WMMA fragment order; all LDS addresses loop-invariant;
// global pointers advance by a fixed step per K-tile (immediate-offset b128s).
// Block: 256 threads (8 wave32), tile 128x128, K-step 32; wave = 32x64 patch.
// ---------------------------------------------------------------------------
template<int AMODE, int BMODE>
__global__ __launch_bounds__(256)
void gemm_kernel(const float* __restrict__ A, long sAb,
                 const float* __restrict__ B, long sBb,
                 const float* __restrict__ Cin, long sCib,
                 float* __restrict__ Cout, long sCob,
                 int N, int K,
                 float alpha, float beta, const float* __restrict__ beta_dev)
{
    __shared__ __align__(32) unsigned short As[128 * 32];
    __shared__ __align__(32) unsigned short Bs[128 * 32];

    const int b = blockIdx.z;
    A += (long)b * sAb;
    B += (long)b * sBb;
    if (Cin) Cin += (long)b * sCib;
    Cout += (long)b * sCob;

    const int tileM = blockIdx.y * 128;
    const int tileN = blockIdx.x * 128;
    const int tid  = threadIdx.x;
    const int wave = tid >> 5;
    const int lane = tid & 31;
    const int wm = (wave & 3) * 32;     // wave row offset in tile
    const int wn = (wave >> 2) * 64;    // wave col offset in tile
    const int half = lane >> 4;
    const int lr   = lane & 15;

    // ---- loop-invariant staging geometry ----
    const float* pA;
    const float* pB;
    unsigned long long* aDst64 = nullptr;
    unsigned long long* bDst64 = nullptr;
    unsigned short* aRow = nullptr;
    unsigned short* bRow = nullptr;
    int apos[4], bpos[4];

    if (AMODE == 0) {
        int am  = tid >> 3;              // row 0..31 (it adds 32 each)
        int akk = (tid & 7) << 2;        // k quad
        pA = A + (long)(tileM + am) * LD + akk;
        aDst64 = (unsigned long long*)&As[am * 32 + fragslot(akk)];
    } else {
        int am  = (tid & 31) << 2;       // m quad
        int akb = tid >> 5;              // k base 0..7 (it adds 8 each)
        pA = A + (long)akb * LD + tileM + am;
        aRow = &As[am * 32];
        #pragma unroll
        for (int it = 0; it < 4; ++it) apos[it] = fragslot(akb + it * 8);
    }
    if (BMODE == 0) {
        int bn  = tid >> 3;
        int bkk = (tid & 7) << 2;
        pB = B + (long)(tileN + bn) * LD + bkk;
        bDst64 = (unsigned long long*)&Bs[bn * 32 + fragslot(bkk)];
    } else {
        int bn  = (tid & 31) << 2;
        int bkb = tid >> 5;
        pB = B + (long)bkb * LD + tileN + bn;
        bRow = &Bs[bn * 32];
        #pragma unroll
        for (int it = 0; it < 4; ++it) bpos[it] = fragslot(bkb + it * 8);
    }
    const long stepA = (AMODE == 0) ? 32 : 32 * LD;
    const long stepB = (BMODE == 0) ? 32 : 32 * LD;

    // loop-invariant fragment addresses (contiguous 32B per lane)
    const v16bf* aF0 = (const v16bf*)&As[(wm +  0 + lr) * 32 + half * 16];
    const v16bf* aF1 = (const v16bf*)&As[(wm + 16 + lr) * 32 + half * 16];
    const v16bf* bF0 = (const v16bf*)&Bs[(wn +  0 + lr) * 32 + half * 16];
    const v16bf* bF1 = (const v16bf*)&Bs[(wn + 16 + lr) * 32 + half * 16];
    const v16bf* bF2 = (const v16bf*)&Bs[(wn + 32 + lr) * 32 + half * 16];
    const v16bf* bF3 = (const v16bf*)&Bs[(wn + 48 + lr) * 32 + half * 16];

    v8f acc[2][4] = {};

    for (int k0 = 0; k0 < K; k0 += 32) {
        // ---- stage A tile ----
        if (AMODE == 0) {
            #pragma unroll
            for (int it = 0; it < 4; ++it) {
                const float4 f = *(const float4*)(pA + it * (32 * LD));
                aDst64[it * 256] = pk4bf(f.x, f.y, f.z, f.w);   // 32 rows x 32 u16 = 256 u64
            }
        } else {
            #pragma unroll
            for (int it = 0; it < 4; ++it) {
                const float4 f = *(const float4*)(pA + it * (8 * LD));
                int p = apos[it];
                aRow[p]      = f2bf(f.x);
                aRow[32 + p] = f2bf(f.y);
                aRow[64 + p] = f2bf(f.z);
                aRow[96 + p] = f2bf(f.w);
            }
        }
        // ---- stage B tile ----
        if (BMODE == 0) {
            #pragma unroll
            for (int it = 0; it < 4; ++it) {
                const float4 f = *(const float4*)(pB + it * (32 * LD));
                bDst64[it * 256] = pk4bf(f.x, f.y, f.z, f.w);
            }
        } else {
            #pragma unroll
            for (int it = 0; it < 4; ++it) {
                const float4 f = *(const float4*)(pB + it * (8 * LD));
                int p = bpos[it];
                bRow[p]      = f2bf(f.x);
                bRow[32 + p] = f2bf(f.y);
                bRow[64 + p] = f2bf(f.z);
                bRow[96 + p] = f2bf(f.w);
            }
        }
        if (k0 + 32 < K) {   // hint next K tile -> global_prefetch_b8
            __builtin_prefetch(pA + stepA, 0, 1);
            __builtin_prefetch(pB + stepB, 0, 1);
        }
        __syncthreads();

        const v16bf a0 = *aF0, a1 = *aF1;
        const v16bf b0 = *bF0, b1 = *bF1, b2 = *bF2, b3 = *bF3;

        acc[0][0] = __builtin_amdgcn_wmma_f32_16x16x32_bf16(false, a0, false, b0, (short)0, acc[0][0], false, false);
        acc[0][1] = __builtin_amdgcn_wmma_f32_16x16x32_bf16(false, a0, false, b1, (short)0, acc[0][1], false, false);
        acc[0][2] = __builtin_amdgcn_wmma_f32_16x16x32_bf16(false, a0, false, b2, (short)0, acc[0][2], false, false);
        acc[0][3] = __builtin_amdgcn_wmma_f32_16x16x32_bf16(false, a0, false, b3, (short)0, acc[0][3], false, false);
        acc[1][0] = __builtin_amdgcn_wmma_f32_16x16x32_bf16(false, a1, false, b0, (short)0, acc[1][0], false, false);
        acc[1][1] = __builtin_amdgcn_wmma_f32_16x16x32_bf16(false, a1, false, b1, (short)0, acc[1][1], false, false);
        acc[1][2] = __builtin_amdgcn_wmma_f32_16x16x32_bf16(false, a1, false, b2, (short)0, acc[1][2], false, false);
        acc[1][3] = __builtin_amdgcn_wmma_f32_16x16x32_bf16(false, a1, false, b3, (short)0, acc[1][3], false, false);
        __syncthreads();

        pA += stepA;
        pB += stepB;
    }

    float ea = alpha;
    float eb = beta * (beta_dev ? *beta_dev : 1.0f);

    // C/D f32 layout: VGPR r -> row = r + 8*half, col = lane&15 (within 16x16)
    #pragma unroll
    for (int mt = 0; mt < 2; ++mt)
        #pragma unroll
        for (int nt = 0; nt < 4; ++nt)
            #pragma unroll
            for (int r = 0; r < 8; ++r) {
                int row = tileM + wm + mt * 16 + r + 8 * half;
                int col = tileN + wn + nt * 16 + lr;
                float v = ea * acc[mt][nt][r];
                if (Cin) v += eb * Cin[(long)row * N + col];
                Cout[(long)row * N + col] = v;
            }
}

// ---------------- small fp32 helper kernels ----------------

// row norms: norm[b,r] = ||w[b,r,:]||  (rows=512, cols=512)
__global__ __launch_bounds__(256)
void rownorm_kernel(const float* __restrict__ w, float* __restrict__ nrm) {
    int r = blockIdx.x, h = blockIdx.y, t = threadIdx.x;
    const float* p = w + ((long)h * 512 + r) * 512;
    float s = 0.f;
    for (int j = t; j < 512; j += 256) { float v = p[j]; s += v * v; }
    __shared__ float sm[256];
    sm[t] = s; __syncthreads();
    for (int k = 128; k > 0; k >>= 1) { if (t < k) sm[t] += sm[t + k]; __syncthreads(); }
    if (t == 0) nrm[(long)h * 512 + r] = sqrtf(sm[0]);
}

// wc[b,r,:] = wm[b,r,:] / (||wm[b,r,:]|| + 1e-5) * nrm0[b,r]
__global__ __launch_bounds__(256)
void renorm_kernel(float* __restrict__ wc, const float* __restrict__ wm,
                   const float* __restrict__ nrm0) {
    int r = blockIdx.x, h = blockIdx.y, t = threadIdx.x;
    const float* pm = wm + ((long)h * 512 + r) * 512;
    float* po = wc + ((long)h * 512 + r) * 512;
    float s = 0.f;
    for (int j = t; j < 512; j += 256) { float v = pm[j]; s += v * v; }
    __shared__ float sm[256];
    sm[t] = s; __syncthreads();
    for (int k = 128; k > 0; k >>= 1) { if (t < k) sm[t] += sm[t + k]; __syncthreads(); }
    float scale = nrm0[(long)h * 512 + r] / (sqrtf(sm[0]) + 1e-5f);
    for (int j = t; j < 512; j += 256) po[j] = pm[j] * scale;
}

// per-chunk, per-head scalars: m_i = mean(momentum), ml_i = sum(muon)/CS (== mean)
__global__ __launch_bounds__(256)
void chunk_scalars_kernel(const float* __restrict__ mom, const float* __restrict__ mlr,
                          float* __restrict__ m_i, float* __restrict__ ml_i) {
    int h = blockIdx.x, t = threadIdx.x;
    __shared__ float sm[256], sl[256];
    sm[t] = mom[(long)h * TSEQ + t];
    sl[t] = mlr[(long)h * TSEQ + t];
    __syncthreads();
    for (int k = 128; k > 0; k >>= 1) { if (t < k) { sm[t] += sm[t + k]; sl[t] += sl[t + k]; } __syncthreads(); }
    if (t == 0) { m_i[h] = sm[0] / (float)CS; ml_i[h] = sl[0] / (float)CS; }
}

// Frobenius inverse norm per head: out[h] = 1/(||X[h]||_F + 1e-7)
__global__ __launch_bounds__(256)
void frob_inv_kernel(const float* __restrict__ X, float* __restrict__ out) {
    int h = blockIdx.x, t = threadIdx.x;
    const float* p = X + (long)h * MSZ;
    float s = 0.f;
    for (long i = t; i < MSZ; i += 256) { float v = p[i]; s += v * v; }
    __shared__ float sm[256];
    sm[t] = s; __syncthreads();
    for (int k = 128; k > 0; k >>= 1) { if (t < k) sm[t] += sm[t + k]; __syncthreads(); }
    if (t == 0) out[h] = 1.0f / (sqrtf(sm[0]) + 1e-7f);
}

// X = G * s[h]
__global__ __launch_bounds__(256)
void scale_kernel(float* __restrict__ X, const float* __restrict__ G,
                  const float* __restrict__ s) {
    int h = blockIdx.y;
    long i = (long)blockIdx.x * 256 + threadIdx.x;
    long o = (long)h * MSZ + i;
    X[o] = G[o] * s[h];
}

// W += X * s[h]
__global__ __launch_bounds__(256)
void axpy_kernel(float* __restrict__ W, const float* __restrict__ X,
                 const float* __restrict__ s) {
    int h = blockIdx.y;
    long i = (long)blockIdx.x * 256 + threadIdx.x;
    long o = (long)h * MSZ + i;
    W[o] += X[o] * s[h];
}

// forward elementwise: S = silu(Gq) * Hq     (shape [BH, CS, DI])
__global__ __launch_bounds__(256)
void fwd_elem_kernel(float* __restrict__ S, const float* __restrict__ Gq,
                     const float* __restrict__ Hq) {
    int h = blockIdx.y;
    long i = (long)blockIdx.x * 256 + threadIdx.x;
    long o = (long)h * CSZ + i;
    float g = Gq[o];
    S[o] = g * sigm(g) * Hq[o];
}

// backward elementwise fusion over [CS, DI] (DI == DD so one index works for k too)
__global__ __launch_bounds__(256)
void bwd_elem_kernel(const float* __restrict__ Gk, const float* __restrict__ Hk,
                     const float* __restrict__ dhT, const float* __restrict__ kc,
                     const float* __restrict__ l0, const float* __restrict__ l1,
                     const float* __restrict__ l2,
                     float* __restrict__ hidL1, float* __restrict__ dgbaT,
                     float* __restrict__ dhbmT, float* __restrict__ kl0,
                     float* __restrict__ kl2) {
    int h = blockIdx.y;
    long i = (long)blockIdx.x * 256 + threadIdx.x;   // 0 .. CSZ-1
    int t = (int)(i / DI);
    int j = (int)(i % DI);
    long o = (long)h * CSZ + i;
    float g  = Gk[o];
    float hb = Hk[o];
    float dh = dhT[o];
    float sg = sigm(g);
    float si = g * sg;                       // silu(g)
    hidL1[o] = si * hb * l1[(long)h * TSEQ + t];
    dhbmT[o] = dh * si;
    float dgate = dh * hb;
    dgbaT[o] = dgate * sg * (1.0f + g * (1.0f - sg));   // silu backprop
    float kv = kc[(long)h * TSEQ * DD + (long)t * DD + j];
    kl0[o] = kv * l0[(long)h * TSEQ + t];
    kl2[o] = kv * l2[(long)h * TSEQ + t];
}

// ---------------- host-side orchestration ----------------

template<int AM, int BM>
static inline void launch_gemm(hipStream_t st, int M, int N, int K,
                               const float* A, long sAb,
                               const float* B, long sBb,
                               const float* Cin, long sCib,
                               float* Cout, long sCob,
                               float alpha, float beta, const float* beta_dev) {
    dim3 grid(N / 128, M / 128, BH);
    gemm_kernel<AM, BM><<<grid, 256, 0, st>>>(A, sAb, B, sBb, Cin, sCib,
                                              Cout, sCob, N, K,
                                              alpha, beta, beta_dev);
}

extern "C" void kernel_launch(void* const* d_in, const int* in_sizes, int n_in,
                              void* d_out, int out_size, void* d_ws, size_t ws_size,
                              hipStream_t stream) {
    (void)in_sizes; (void)n_in; (void)out_size; (void)ws_size;
    const float* w0  = (const float*)d_in[0];   // [BH, DI, DD]
    const float* w1  = (const float*)d_in[1];   // [BH, DD, DI]
    const float* w2  = (const float*)d_in[2];   // [BH, DI, DD]
    const float* q   = (const float*)d_in[3];   // [BH, T, DD]
    const float* k   = (const float*)d_in[4];
    const float* v   = (const float*)d_in[5];
    const float* lr0 = (const float*)d_in[6];   // [BH, T, 1]
    const float* lr1 = (const float*)d_in[7];
    const float* lr2 = (const float*)d_in[8];
    const float* mom = (const float*)d_in[9];
    const float* mlr = (const float*)d_in[10];
    float* out = (float*)d_out;                 // [BH, T, DD]

    // ---- workspace carve-up (fp32) ----
    float* ws = (float*)d_ws;
    auto take = [&](long n) { float* p = ws; ws += n; return p; };
    float* w0m = take((long)BH * MSZ);
    float* w1m = take((long)BH * MSZ);
    float* w2m = take((long)BH * MSZ);
    float* w0c = take((long)BH * MSZ);
    float* w1c = take((long)BH * MSZ);
    float* w2c = take((long)BH * MSZ);
    float* dm0 = take((long)BH * MSZ);   // dm0..dm2 contiguous for one memset
    float* dm1 = take((long)BH * MSZ);
    float* dm2 = take((long)BH * MSZ);
    float* nsX = take((long)BH * MSZ);
    float* nsA = take((long)BH * MSZ);
    float* nsB = take((long)BH * MSZ);
    float* Hq    = take((long)BH * CSZ);
    float* Gq    = take((long)BH * CSZ);
    float* Smat  = take((long)BH * CSZ);
    float* Gk    = take((long)BH * CSZ);
    float* Hk    = take((long)BH * CSZ);
    float* dhT   = take((long)BH * CSZ);
    float* hidL1 = take((long)BH * CSZ);
    float* dgbaT = take((long)BH * CSZ);
    float* dhbmT = take((long)BH * CSZ);
    float* kl0   = take((long)BH * CSZ);
    float* kl2   = take((long)BH * CSZ);
    float* w0n = take((long)BH * 512);
    float* w1n = take((long)BH * 512);
    float* w2n = take((long)BH * 512);
    float* m_i  = take(BH);
    float* ml_i = take(BH);
    float* fro  = take(BH);

    const long wbytes = (long)BH * MSZ * sizeof(float);

    // ---- init: carry0 = (w, w, w, 0,0,0) ; initial row norms ----
    hipMemcpyAsync(w0m, w0, wbytes, hipMemcpyDeviceToDevice, stream);
    hipMemcpyAsync(w1m, w1, wbytes, hipMemcpyDeviceToDevice, stream);
    hipMemcpyAsync(w2m, w2, wbytes, hipMemcpyDeviceToDevice, stream);
    hipMemcpyAsync(w0c, w0, wbytes, hipMemcpyDeviceToDevice, stream);
    hipMemcpyAsync(w1c, w1, wbytes, hipMemcpyDeviceToDevice, stream);
    hipMemcpyAsync(w2c, w2, wbytes, hipMemcpyDeviceToDevice, stream);
    hipMemsetAsync(dm0, 0, 3 * wbytes, stream);
    rownorm_kernel<<<dim3(512, BH), 256, 0, stream>>>(w0, w0n);
    rownorm_kernel<<<dim3(512, BH), 256, 0, stream>>>(w1, w1n);
    rownorm_kernel<<<dim3(512, BH), 256, 0, stream>>>(w2, w2n);

    const dim3 egrid(CSZ / 256, BH);
    const dim3 mgrid(MSZ / 256, BH);

    float* dms[3]  = {dm0, dm1, dm2};
    float* wms[3]  = {w0m, w1m, w2m};
    float* wcs[3]  = {w0c, w1c, w2c};
    float* nrms[3] = {w0n, w1n, w2n};

    for (int i = 0; i < NCH; ++i) {
        const float* qc = q + (long)i * CS * DD;     // head stride T*DD
        const float* kc = k + (long)i * CS * DD;
        const float* vc = v + (long)i * CS * DD;

        chunk_scalars_kernel<<<BH, 256, 0, stream>>>(mom + (long)i * CS,
                                                     mlr + (long)i * CS, m_i, ml_i);

        // ---- forward with current normalized weights ----
        // Hq = qc @ w2c^T ; Gq = qc @ w0c^T      [CS, DI]
        launch_gemm<0,0>(stream, CS, DI, DD, qc, (long)TSEQ * DD,
                         w2c, MSZ, nullptr, 0, Hq, CSZ, 1.f, 0.f, nullptr);
        launch_gemm<0,0>(stream, CS, DI, DD, qc, (long)TSEQ * DD,
                         w0c, MSZ, nullptr, 0, Gq, CSZ, 1.f, 0.f, nullptr);
        fwd_elem_kernel<<<egrid, 256, 0, stream>>>(Smat, Gq, Hq);
        // out_chunk = S @ w1c^T  ->  out[:, i*CS:(i+1)*CS, :]
        launch_gemm<0,0>(stream, CS, DD, DI, Smat, CSZ,
                         w1c, MSZ, nullptr, 0,
                         out + (long)i * CS * DD, (long)TSEQ * DD, 1.f, 0.f, nullptr);

        // ---- backward (fast-weight grads) ----
        launch_gemm<0,0>(stream, CS, DI, DD, kc, (long)TSEQ * DD,
                         w0c, MSZ, nullptr, 0, Gk, CSZ, 1.f, 0.f, nullptr);
        launch_gemm<0,0>(stream, CS, DI, DD, kc, (long)TSEQ * DD,
                         w2c, MSZ, nullptr, 0, Hk, CSZ, 1.f, 0.f, nullptr);
        // dhidden^T = vc @ w1c   (B row-major -> BMODE 1)
        launch_gemm<0,1>(stream, CS, DI, DD, vc, (long)TSEQ * DD,
                         w1c, MSZ, nullptr, 0, dhT, CSZ, 1.f, 0.f, nullptr);
        bwd_elem_kernel<<<egrid, 256, 0, stream>>>(Gk, Hk, dhT, kc,
                                                   lr0 + (long)i * CS,
                                                   lr1 + (long)i * CS,
                                                   lr2 + (long)i * CS,
                                                   hidL1, dgbaT, dhbmT, kl0, kl2);
        // dm1 = vc^T @ hidL1 + m_i*dm1   [DD, DI]
        launch_gemm<1,1>(stream, DD, DI, CS, vc, (long)TSEQ * DD,
                         hidL1, CSZ, dm1, MSZ, dm1, MSZ, 1.f, 1.f, m_i);
        // dm0 = dgbaT^T @ kl0 + m_i*dm0  [DI, DD]
        launch_gemm<1,1>(stream, DI, DD, CS, dgbaT, CSZ,
                         kl0, CSZ, dm0, MSZ, dm0, MSZ, 1.f, 1.f, m_i);
        // dm2 = dhbmT^T @ kl2 + m_i*dm2  [DI, DD]
        launch_gemm<1,1>(stream, DI, DD, CS, dhbmT, CSZ,
                         kl2, CSZ, dm2, MSZ, dm2, MSZ, 1.f, 1.f, m_i);

        // ---- Muon NS5 + update + renorm, per fast weight ----
        for (int m = 0; m < 3; ++m) {
            frob_inv_kernel<<<BH, 256, 0, stream>>>(dms[m], fro);
            scale_kernel<<<mgrid, 256, 0, stream>>>(nsX, dms[m], fro);
            float* X = nsX; float* Abuf = nsA; float* Bbuf = nsB;
            for (int s = 0; s < 5; ++s) {
                // A = X @ X^T   (B = X with k fast along columns -> BMODE 0)
                launch_gemm<0,0>(stream, 512, 512, 512, X, MSZ,
                                 X, MSZ, nullptr, 0, Abuf, MSZ, 1.f, 0.f, nullptr);
                // B = c*(A@A) + b*A   (B row-major -> BMODE 1)
                launch_gemm<0,1>(stream, 512, 512, 512, Abuf, MSZ,
                                 Abuf, MSZ, Abuf, MSZ, Bbuf, MSZ,
                                 NSC[s][2], NSC[s][1], nullptr);
                // Xnew = B@X + a*X   (write into Abuf, then swap)
                launch_gemm<0,1>(stream, 512, 512, 512, Bbuf, MSZ,
                                 X, MSZ, X, MSZ, Abuf, MSZ,
                                 1.f, NSC[s][0], nullptr);
                float* t2 = X; X = Abuf; Abuf = t2;
            }
            axpy_kernel<<<mgrid, 256, 0, stream>>>(wms[m], X, ml_i);
            renorm_kernel<<<dim3(512, BH), 256, 0, stream>>>(wcs[m], wms[m], nrms[m]);
        }
    }

    // ---- final forward-only chunk with final normalized weights ----
    {
        const float* qc = q + (long)NCH * CS * DD;
        launch_gemm<0,0>(stream, CS, DI, DD, qc, (long)TSEQ * DD,
                         w2c, MSZ, nullptr, 0, Hq, CSZ, 1.f, 0.f, nullptr);
        launch_gemm<0,0>(stream, CS, DI, DD, qc, (long)TSEQ * DD,
                         w0c, MSZ, nullptr, 0, Gq, CSZ, 1.f, 0.f, nullptr);
        fwd_elem_kernel<<<egrid, 256, 0, stream>>>(Smat, Gq, Hq);
        launch_gemm<0,0>(stream, CS, DD, DI, Smat, CSZ,
                         w1c, MSZ, nullptr, 0,
                         out + (long)NCH * CS * DD, (long)TSEQ * DD, 1.f, 0.f, nullptr);
    }
}
